// CRF_45973329937127
// MI455X (gfx1250) — matrix operand order repaired and checked by reading, not compiled, not run on hardware
//
#include <hip/hip_runtime.h>
#include <hip/hip_bf16.h>

typedef __attribute__((ext_vector_type(16))) _Float16 v16h;
typedef __attribute__((ext_vector_type(8)))  float    v8f;

#define Bsz     256
#define Tn      1024
#define Kn      128
#define MB      16          // batches per workgroup
#define THREADS 256         // 8 waves of 32
#define NEG_INF (-3.402823466e+38f)

// ---------------------------------------------------------------------------
// Forward CRF: one workgroup owns MB=16 batch rows for the whole T scan.
// Per step:  w_j = e_j + log( sum_i exp(alphaN_i) * exp(trans_ij) )
// as a 16x128 @ 128x128 GEMM in split-f16 WMMA (hi/lo error compensation).
// B fragments (exp(trans)) are register-resident for the whole scan; alphaN
// and the running log-norm are register-resident per thread; per-batch
// reductions are 16-lane shuffle butterflies -> only 2 barriers per step.
// ---------------------------------------------------------------------------
__global__ __launch_bounds__(THREADS)
void crf_fwd_45973_kernel(const float* __restrict__ inputs,
                          const float* __restrict__ trans,
                          const int*   __restrict__ tags,
                          const int*   __restrict__ mask,
                          float*       __restrict__ loglik)
{
    // exp(trans) staged once in B-fragment layout (hi/lo): [jt][kc][lane][half]
    __shared__ __align__(32) _Float16 etFragH[8][4][32][16];   // 32 KB
    __shared__ __align__(32) _Float16 etFragL[8][4][32][16];   // 32 KB
    // exp(alphaN) in A-fragment layout (hi/lo): [kc][lane][half]
    __shared__ __align__(32) _Float16 aFragH[4][32][16];       // 4 KB
    __shared__ __align__(32) _Float16 aFragL[4][32][16];       // 4 KB
    __shared__ float wNew[MB][Kn];                             // 8 KB

    const int tid   = threadIdx.x;
    const int lane  = tid & 31;
    const int wave  = tid >> 5;          // 0..7 -> j-tile owner
    const int bloc  = tid >> 4;          // 0..15 local batch row
    const int sub   = tid & 15;
    const int bbase = blockIdx.x * MB;
    const int bglob = bbase + bloc;

    // ---- build exp(transitions) fragments in LDS (init-only) ----
    for (int idx = tid; idx < 8 * 4 * 32 * 16; idx += THREADS) {
        int h    = idx & 15;
        int ln   = (idx >> 4) & 31;
        int kc   = (idx >> 9) & 3;
        int jt0  = idx >> 11;
        int krow = kc * 32 + ((ln < 16) ? 0 : 16) + h;   // contiguous k order chosen here (matches A): K = base + half
        int n    = jt0 * 16 + (ln & 15);
        float v  = __expf(trans[krow * Kn + n]);
        _Float16 hi = (_Float16)v;
        etFragH[jt0][kc][ln][h] = hi;
        etFragL[jt0][kc][ln][h] = (_Float16)(v - (float)hi);
    }

    // ---- register state: normalized alpha (this thread's 8 states) + log-norm
    float aN[8];
    float accR;
    {
        float pm = NEG_INF;
        float e0v[8];
        #pragma unroll
        for (int u = 0; u < 8; ++u) {
            e0v[u] = inputs[((size_t)bglob * Tn) * Kn + sub + 16 * u];
            pm = fmaxf(pm, e0v[u]);
        }
        pm = fmaxf(pm, __shfl_xor(pm, 8));
        pm = fmaxf(pm, __shfl_xor(pm, 4));
        pm = fmaxf(pm, __shfl_xor(pm, 2));
        pm = fmaxf(pm, __shfl_xor(pm, 1));
        accR = pm;
        #pragma unroll
        for (int u = 0; u < 8; ++u) aN[u] = e0v[u] - pm;
    }
    __syncthreads();

    // ---- per-lane WMMA coordinates + register-resident B fragments ----
    const int jt   = wave;
    const int nloc = lane & 15;
    const int mrow = (lane < 16) ? 0 : 8;       // M = r + mrow for C/D VGPR r

    v16h bhReg[4], blReg[4];
    #pragma unroll
    for (int kc = 0; kc < 4; ++kc) {
        bhReg[kc] = *(const v16h*)(&etFragH[jt][kc][lane][0]);
        blReg[kc] = *(const v16h*)(&etFragL[jt][kc][lane][0]);
    }

    float eReg[8];
    #pragma unroll
    for (int r = 0; r < 8; ++r) {
        int M = r + mrow;
        eReg[r] = inputs[(((size_t)(bbase + M)) * Tn + 1) * Kn + jt * 16 + nloc];
    }

    for (int t = 1; t < Tn; ++t) {
        // -- phase 1: P = exp(aN) into A-fragment layout (split hi/lo)
        #pragma unroll
        for (int u = 0; u < 8; ++u) {
            int i  = sub + 16 * u;
            float p = __expf(aN[u]);
            int kc = i >> 5;
            int k2 = i & 31;
            // contiguous k order used here (matches B): K' in [0..15] -> lanes 0-15, [16..31] -> lanes 16-31
            int ln2 = bloc + ((k2 >> 4) << 4);
            int h   = k2 & 15;
            _Float16 ph = (_Float16)p;
            aFragH[kc][ln2][h] = ph;
            aFragL[kc][ln2][h] = (_Float16)(p - (float)ph);
        }
        __syncthreads();   // B1: aFrag ready

        // -- phase 2: 3 independent WMMA chains (hi*hi, hi*lo, lo*hi)
        v8f c0 = {}, c1 = {}, c2 = {};
        #pragma unroll
        for (int kc = 0; kc < 4; ++kc) {
            v16h ah = *(const v16h*)(&aFragH[kc][lane][0]);
            v16h al = *(const v16h*)(&aFragL[kc][lane][0]);
            c0 = __builtin_amdgcn_wmma_f32_16x16x32_f16(false, ah, false, bhReg[kc], (short)0, c0, false, false);
            c1 = __builtin_amdgcn_wmma_f32_16x16x32_f16(false, ah, false, blReg[kc], (short)0, c1, false, false);
            c2 = __builtin_amdgcn_wmma_f32_16x16x32_f16(false, al, false, bhReg[kc], (short)0, c2, false, false);
        }

        // prefetch t+2 emission rows; load t+1 into regs (consumed next iter)
        if (t + 2 < Tn) {
            #pragma unroll
            for (int r = 0; r < 8; ++r) {
                int M = r + mrow;
                __builtin_prefetch(&inputs[(((size_t)(bbase + M)) * Tn + (t + 2)) * Kn + jt * 16 + nloc], 0, 1);
            }
        }
        float eNext[8];
        #pragma unroll
        for (int r = 0; r < 8; ++r) eNext[r] = 0.0f;
        if (t + 1 < Tn) {
            #pragma unroll
            for (int r = 0; r < 8; ++r) {
                int M = r + mrow;
                eNext[r] = inputs[(((size_t)(bbase + M)) * Tn + (t + 1)) * Kn + jt * 16 + nloc];
            }
        }

        #pragma unroll
        for (int r = 0; r < 8; ++r) {
            int M = r + mrow;
            float v = (c0[r] + c1[r]) + c2[r];
            wNew[M][jt * 16 + nloc] = eReg[r] + __logf(v);
        }
        __syncthreads();   // B2: wNew ready

        // -- phase 3: per-batch max via 16-lane butterfly, masked reg update
        float wv[8];
        float pm = NEG_INF;
        #pragma unroll
        for (int u = 0; u < 8; ++u) {
            wv[u] = wNew[bloc][sub + 16 * u];
            pm = fmaxf(pm, wv[u]);
        }
        pm = fmaxf(pm, __shfl_xor(pm, 8));
        pm = fmaxf(pm, __shfl_xor(pm, 4));
        pm = fmaxf(pm, __shfl_xor(pm, 2));
        pm = fmaxf(pm, __shfl_xor(pm, 1));
        const int mi = mask[(size_t)bglob * Tn + t];
        if (mi) {
            accR += pm;
            #pragma unroll
            for (int u = 0; u < 8; ++u) aN[u] = wv[u] - pm;
        }
        #pragma unroll
        for (int r = 0; r < 8; ++r) eReg[r] = eNext[r];
        // no barrier needed: next phase1 writes aFrag (reads finished before B2)
        // and reads only this thread's registers.
    }

    // ---- log denominator: accR + log(sum exp(aN)) via butterfly sum ----
    float den;
    {
        float s = 0.0f;
        #pragma unroll
        for (int u = 0; u < 8; ++u) s += __expf(aN[u]);
        s += __shfl_xor(s, 8);
        s += __shfl_xor(s, 4);
        s += __shfl_xor(s, 2);
        s += __shfl_xor(s, 1);
        den = accR + __logf(s);
    }

    // ---- joint (numerator): gathers + butterfly reduce, write per-batch loglik
    {
        float part = 0.0f;
        int   cnt  = 0;
        const size_t tb = (size_t)bglob * Tn;
        for (int t = sub; t < Tn; t += 16) {
            int m_t = mask[tb + t];
            cnt += m_t;
            if (t < Tn - 1) {
                float mfn = (float)mask[tb + t + 1];
                int tg  = tags[tb + t];
                int tgn = tags[tb + t + 1];
                part += trans[tg * Kn + tgn] * mfn;
                part += inputs[(tb + t) * Kn + tg] * (float)m_t;
            }
        }
        part += __shfl_xor(part, 8);
        part += __shfl_xor(part, 4);
        part += __shfl_xor(part, 2);
        part += __shfl_xor(part, 1);
        cnt += __shfl_xor(cnt, 8);
        cnt += __shfl_xor(cnt, 4);
        cnt += __shfl_xor(cnt, 2);
        cnt += __shfl_xor(cnt, 1);
        if (sub == 0) {
            int last_idx = cnt - 1;
            if (last_idx < 0)      last_idx = 0;
            if (last_idx > Tn - 1) last_idx = Tn - 1;
            int last_tag = tags[tb + last_idx];
            float num = part + inputs[(tb + (Tn - 1)) * Kn + last_tag] * (float)mask[tb + Tn - 1];
            loglik[bglob] = num - den;
        }
    }
}

// ---------------------------------------------------------------------------
// Deterministic fixed-order tree reduction of the 256 per-batch logliks.
// ---------------------------------------------------------------------------
__global__ __launch_bounds__(256)
void crf_reduce_45973_kernel(const float* __restrict__ v, float* __restrict__ out)
{
    __shared__ float red[256];
    int tid = threadIdx.x;
    red[tid] = v[tid];
    __syncthreads();
    for (int s = 128; s > 0; s >>= 1) {
        if (tid < s) red[tid] += red[tid + s];
        __syncthreads();
    }
    if (tid == 0) out[0] = red[0];
}

extern "C" void kernel_launch(void* const* d_in, const int* in_sizes, int n_in,
                              void* d_out, int out_size, void* d_ws, size_t ws_size,
                              hipStream_t stream)
{
    (void)in_sizes; (void)n_in; (void)out_size; (void)ws_size;
    const float* inputs = (const float*)d_in[0];
    const float* trans  = (const float*)d_in[1];
    const int*   tags   = (const int*)d_in[2];
    const int*   mask   = (const int*)d_in[3];
    float* out = (float*)d_out;
    float* ws  = (float*)d_ws;   // 256 per-batch logliks

    crf_fwd_45973_kernel<<<Bsz / MB, THREADS, 0, stream>>>(inputs, trans, tags, mask, ws);
    crf_reduce_45973_kernel<<<1, 256, 0, stream>>>(ws, out);
}
